// OnlineKNN_37967510897033
// MI455X (gfx1250) — matrix operand order repaired and checked by reading, compiled, not run on hardware
//
#include <hip/hip_runtime.h>
#include <hip/hip_bf16.h>

// ---------------------------------------------------------------------------
// Online-KNN for MI455X (gfx1250, wave32, WMMA).
//   sim = F[512,256] @ Q[131072,256]^T ; top-200/row ; exp(sim/0.07) class vote.
//
// Pass A (screen): bf16 WMMA (V_WMMA_F32_16X16X32_BF16, 8192 MACs/instr) builds
//   a per-row histogram of approximate sims.  bf16 error bound: 2^-7 * ||a||*||b||
//   = 0.0078 for unit vectors, so lowering the rank-200 bucket edge by 0.009
//   provably captures every true top-200 neighbor.
// Pass B (exact): native f32 WMMA (V_WMMA_F32_16X16X4_F32) recomputes sims and
//   appends all candidates above the conservative threshold.
// Finalize: per-row exact top-200 by 32-bit-key binary search, exp weights,
//   class vote, argmax (first-index tie break), accuracy.
//
// Queue (134 MB) is read from HBM exactly once per pass; each wave holds its
// 16-column B panel entirely in VGPRs.  Features stay hot in the 192 MB L2.
// ---------------------------------------------------------------------------

typedef __attribute__((ext_vector_type(2))) float v2f;
typedef __attribute__((ext_vector_type(8))) float v8f;
typedef __attribute__((ext_vector_type(8))) __bf16 v8bf;
typedef __attribute__((ext_vector_type(16))) __bf16 v16bf;

#define KNN_N 512
#define KNN_D 256
#define KNN_K 131072
#define KNN_TOPK 200
#define KNN_NC 1000
#define KNN_INV_T 14.285714285714286f
#define KNN_MARGIN 0.009f     // > 2^-7 bf16 screening error bound
#define NBUCKET 4096
#define CAND_CAP 4096
#define COLS_PER_BLOCK 128    // 8 waves x 16 cols

__device__ __forceinline__ unsigned f32_key(float s) {
    unsigned u = __float_as_uint(s);
    return (u & 0x80000000u) ? ~u : (u | 0x80000000u);
}
__device__ __forceinline__ float key_f32(unsigned k) {
    unsigned u = (k & 0x80000000u) ? (k ^ 0x80000000u) : ~k;
    return __uint_as_float(u);
}
__device__ __forceinline__ v16bf cvt16(const float* __restrict__ p) {
    v16bf r;
#pragma unroll
    for (int i = 0; i < 16; ++i) r[i] = (__bf16)p[i];
    return r;
}

// ---------------- zero-init (d_out/d_ws are poisoned by harness) -----------
__global__ void knn_zero(unsigned* __restrict__ hist, int* __restrict__ ccount,
                         float* __restrict__ out) {
    size_t stride = (size_t)gridDim.x * blockDim.x;
    for (size_t i = (size_t)blockIdx.x * blockDim.x + threadIdx.x;
         i < (size_t)KNN_N * NBUCKET; i += stride) hist[i] = 0u;
    for (size_t i = (size_t)blockIdx.x * blockDim.x + threadIdx.x;
         i < (size_t)KNN_N; i += stride) ccount[i] = 0;
    for (size_t i = (size_t)blockIdx.x * blockDim.x + threadIdx.x;
         i < (size_t)(1 + KNN_N * KNN_NC); i += stride) out[i] = 0.0f;
}

// ---------------- features -> bf16 copy (256 KB, L2-resident) --------------
__global__ void knn_prep(const float* __restrict__ feat, __bf16* __restrict__ fhi) {
    int i = blockIdx.x * blockDim.x + threadIdx.x;
    if (i < KNN_N * KNN_D) fhi[i] = (__bf16)feat[i];
}

// ---------------- Pass A: bf16 WMMA screening histogram --------------------
__global__ __launch_bounds__(256) void knn_hist(
    const __bf16* __restrict__ fhi, const float* __restrict__ qfeat,
    unsigned* __restrict__ hist) {
    const int lane  = threadIdx.x & 31;
    const int wave  = threadIdx.x >> 5;
    const int nloc  = lane & 15;   // M row (A) / N col (B,C)
    const int khalf = lane >> 4;   // upper half-wave K group
    const int qcol  = blockIdx.x * COLS_PER_BLOCK + wave * 16 + nloc;

    // B panel (16 cols x 256 D, bf16-hi) in 64 VGPRs; per lane a contiguous
    // K-run of 16: B layout => lanes 0-15 hold K=c*32+0..15, lanes 16-31 +16.
    v16bf Bh[8];
    const float* qrow = qfeat + (size_t)qcol * KNN_D;
#pragma unroll
    for (int c = 0; c < 8; ++c)
        Bh[c] = cvt16(qrow + c * 32 + khalf * 16);

    for (int s = 0; s < 32; ++s) {
        const __bf16* arow = fhi + (size_t)(s * 16 + nloc) * KNN_D;
        if (s + 1 < 32) __builtin_prefetch(arow + 16 * KNN_D, 0, 0);
        v8f acc0 = {}, acc1 = {};
#pragma unroll
        for (int c = 0; c < 8; c += 2) {
            // A layout: elems 0..15 -> K = c*32 + khalf*16 + i (matches B)
            v8bf a00 = *(const v8bf*)(arow + c * 32 + khalf * 16);
            v8bf a01 = *(const v8bf*)(arow + c * 32 + 8 + khalf * 16);
            v16bf A0 = __builtin_shufflevector(a00, a01, 0, 1, 2, 3, 4, 5, 6, 7,
                                               8, 9, 10, 11, 12, 13, 14, 15);
            v8bf a10 = *(const v8bf*)(arow + (c + 1) * 32 + khalf * 16);
            v8bf a11 = *(const v8bf*)(arow + (c + 1) * 32 + 8 + khalf * 16);
            v16bf A1 = __builtin_shufflevector(a10, a11, 0, 1, 2, 3, 4, 5, 6, 7,
                                               8, 9, 10, 11, 12, 13, 14, 15);
            acc0 = __builtin_amdgcn_wmma_f32_16x16x32_bf16(
                false, A0, false, Bh[c], (short)0, acc0, false, false);
            acc1 = __builtin_amdgcn_wmma_f32_16x16x32_bf16(
                false, A1, false, Bh[c + 1], (short)0, acc1, false, false);
        }
        v8f acc = acc0 + acc1;
#pragma unroll
        for (int j = 0; j < 8; ++j) {
            int row = s * 16 + j + khalf * 8;   // C: vgpr j -> M, +8 upper half
            unsigned key = f32_key(acc[j]);
            atomicAdd(&hist[(size_t)row * NBUCKET + (key >> 20)], 1u);
        }
    }
}

// ---------------- conservative per-row key threshold -----------------------
__global__ void knn_thresh(const unsigned* __restrict__ hist,
                           unsigned* __restrict__ tkey) {
    int row = blockIdx.x * blockDim.x + threadIdx.x;
    if (row >= KNN_N) return;
    unsigned cum = 0;
    int b = 0;
    for (int bb = NBUCKET - 1; bb >= 0; --bb) {
        unsigned h = hist[(size_t)row * NBUCKET + bb];
        if (cum + h >= (unsigned)KNN_TOPK) { b = bb; break; }
        cum += h;
    }
    if (b == 0) { tkey[row] = 0u; return; }
    // >=200 approx sims lie at/above this bucket's lower edge; true values are
    // within KNN_MARGIN of approx, so edge - margin under-bounds true s_200.
    float edge = key_f32((unsigned)b << 20);
    tkey[row] = f32_key(edge - KNN_MARGIN);
}

// ---------------- Pass B: exact f32 WMMA, collect candidates ---------------
__global__ __launch_bounds__(256) void knn_collect(
    const float* __restrict__ feat, const float* __restrict__ qfeat,
    const int* __restrict__ qlab, const unsigned* __restrict__ tkey,
    int* __restrict__ ccount, uint2* __restrict__ cand) {
    const int lane  = threadIdx.x & 31;
    const int wave  = threadIdx.x >> 5;
    const int nloc  = lane & 15;
    const int khalf = lane >> 4;
    const int qcol  = blockIdx.x * COLS_PER_BLOCK + wave * 16 + nloc;

    // B panel (16 cols x 256 D, f32) in 128 VGPRs
    v2f Bf[64];
    const float* qrow = qfeat + (size_t)qcol * KNN_D + khalf * 2;
#pragma unroll
    for (int k = 0; k < 64; ++k) Bf[k] = *(const v2f*)(qrow + k * 4);
    const int lab = qlab[qcol];

    for (int s = 0; s < 32; ++s) {
        const float* arow = feat + (size_t)(s * 16 + nloc) * KNN_D + khalf * 2;
        if (s + 1 < 32) __builtin_prefetch(arow + 16 * KNN_D, 0, 0);
        v8f acc0 = {}, acc1 = {};
#pragma unroll
        for (int k = 0; k < 64; k += 2) {
            v2f a0 = *(const v2f*)(arow + k * 4);
            v2f a1 = *(const v2f*)(arow + k * 4 + 4);
            acc0 = __builtin_amdgcn_wmma_f32_16x16x4_f32(
                false, a0, false, Bf[k], (short)0, acc0, false, false);
            acc1 = __builtin_amdgcn_wmma_f32_16x16x4_f32(
                false, a1, false, Bf[k + 1], (short)0, acc1, false, false);
        }
        v8f acc = acc0 + acc1;
#pragma unroll
        for (int j = 0; j < 8; ++j) {
            int row = s * 16 + j + khalf * 8;
            unsigned key = f32_key(acc[j]);
            if (key >= tkey[row]) {
                int i = atomicAdd(&ccount[row], 1);
                if (i < CAND_CAP) {
                    uint2 c; c.x = key; c.y = (unsigned)lab;
                    cand[(size_t)row * CAND_CAP + i] = c;
                }
            }
        }
    }
}

// ---------------- exact top-200 + class vote + argmax + accuracy -----------
__device__ __forceinline__ int block_count_ge(const uint2* __restrict__ C, int cnt,
                                              unsigned t, int tid, int* sc) {
    if (tid == 0) *sc = 0;
    __syncthreads();
    int local = 0;
    for (int i = tid; i < cnt; i += 256)
        if (C[i].x >= t) ++local;
    atomicAdd(sc, local);
    __syncthreads();
    int r = *sc;
    __syncthreads();
    return r;
}

__global__ __launch_bounds__(256) void knn_final(
    const uint2* __restrict__ cand, const int* __restrict__ ccount,
    const int* __restrict__ labels, float* __restrict__ scores,
    float* __restrict__ out_acc) {
    const int row = blockIdx.x;
    const int tid = threadIdx.x;
    int cnt = ccount[row];
    if (cnt > CAND_CAP) cnt = CAND_CAP;
    int need = KNN_TOPK < cnt ? KNN_TOPK : cnt;
    const uint2* C = cand + (size_t)row * CAND_CAP;

    __shared__ int sc;
    __shared__ int take;

    if (need > 0) {
        // binary search the need-th largest 32-bit key among candidates
        unsigned lo = 0u, hi = 0xFFFFFFFFu;
        while (lo < hi) {
            unsigned mid =
                lo + (unsigned)((((unsigned long long)hi - lo + 1ull) >> 1));
            int c = block_count_ge(C, cnt, mid, tid, &sc);
            if (c >= need) lo = mid; else hi = mid - 1;
        }
        unsigned tstar = lo;
        int cgt = (tstar == 0xFFFFFFFFu)
                      ? 0
                      : block_count_ge(C, cnt, tstar + 1, tid, &sc);
        int rem = need - cgt;
        if (tid == 0) take = 0;
        __syncthreads();
        for (int i = tid; i < cnt; i += 256) {
            unsigned key = C[i].x;
            bool add = false;
            if (key > tstar) add = true;
            else if (key == tstar) {
                int t2 = atomicAdd(&take, 1);
                if (t2 < rem) add = true;
            }
            if (add) {
                float sv = key_f32(key);
                atomicAdd(&scores[row * KNN_NC + (int)C[i].y],
                          __expf(sv * KNN_INV_T));
            }
        }
    }
    __syncthreads();
    __threadfence();

    // argmax over classes (first-index tie break, matching jnp.argmax)
    __shared__ float sb[256];
    __shared__ int si[256];
    float best = -1.0f;
    int bidx = KNN_NC;
    for (int c = tid; c < KNN_NC; c += 256) {
        float v = scores[row * KNN_NC + c];
        if (v > best || (v == best && c < bidx)) { best = v; bidx = c; }
    }
    sb[tid] = best; si[tid] = bidx;
    __syncthreads();
    for (int off = 128; off > 0; off >>= 1) {
        if (tid < off) {
            if (sb[tid + off] > sb[tid] ||
                (sb[tid + off] == sb[tid] && si[tid + off] < si[tid])) {
                sb[tid] = sb[tid + off]; si[tid] = si[tid + off];
            }
        }
        __syncthreads();
    }
    if (tid == 0 && si[0] == labels[row])
        atomicAdd(out_acc, 1.0f / (float)KNN_N);
}

// ---------------------------------------------------------------------------
extern "C" void kernel_launch(void* const* d_in, const int* in_sizes, int n_in,
                              void* d_out, int out_size, void* d_ws, size_t ws_size,
                              hipStream_t stream) {
    (void)in_sizes; (void)n_in; (void)out_size; (void)ws_size;
    const float* feat   = (const float*)d_in[0];  // [512,256]
    const int*   labels = (const int*)d_in[1];    // [512]
    const float* qfeat  = (const float*)d_in[2];  // [131072,256]
    const int*   qlab   = (const int*)d_in[3];    // [131072]
    float* out    = (float*)d_out;                // [0]=accuracy, [1..]=pred_scores
    float* scores = out + 1;

    // workspace layout (~24.5 MB)
    char* ws = (char*)d_ws;
    const size_t HIST_BYTES = (size_t)KNN_N * NBUCKET * sizeof(unsigned); // 8 MB
    unsigned* hist   = (unsigned*)ws;
    int*      ccount = (int*)(ws + HIST_BYTES);                 // 2 KB
    unsigned* tkey   = (unsigned*)(ws + HIST_BYTES + 4096);     // 2 KB
    __bf16*   fhi    = (__bf16*)(ws + HIST_BYTES + 16384);      // 256 KB
    uint2*    cand   = (uint2*)(ws + HIST_BYTES + (512 << 10)); // 16 MB

    knn_zero<<<2048, 256, 0, stream>>>(hist, ccount, out);
    knn_prep<<<(KNN_N * KNN_D + 255) / 256, 256, 0, stream>>>(feat, fhi);
    knn_hist<<<KNN_K / COLS_PER_BLOCK, 256, 0, stream>>>(fhi, qfeat, hist);
    knn_thresh<<<(KNN_N + 255) / 256, 256, 0, stream>>>(hist, tkey);
    knn_collect<<<KNN_K / COLS_PER_BLOCK, 256, 0, stream>>>(
        feat, qfeat, qlab, tkey, ccount, cand);
    knn_final<<<KNN_N, 256, 0, stream>>>(cand, ccount, labels, scores, out);
}
